// GNNLinkPredictor_25872882991658
// MI455X (gfx1250) — compile-verified, matrix operand order checked by reading, not run on hardware
//
#include <hip/hip_runtime.h>
#include <hip/hip_bf16.h>

#define N_NODES 50000
#define N_EDGES 800000
#define N_PRED  400000
#define D_IN    64
#define D_HID   128
#define D_OUT   64

typedef __attribute__((ext_vector_type(16))) _Float16      v16h;
typedef __attribute__((ext_vector_type(8)))  float         v8f;
typedef __attribute__((ext_vector_type(4)))  unsigned int  v4u;

union FragH {
    v16h v;
    v4u  q[2];
    _Float16 h[16];
};

// ---------------------------------------------------------------- utilities
__global__ void zero_f32_kernel(float* __restrict__ p, long long n) {
    long long i = (long long)blockIdx.x * blockDim.x + threadIdx.x;
    long long stride = (long long)gridDim.x * blockDim.x;
    for (; i < n; i += stride) p[i] = 0.0f;
}

__global__ void cvt_f32_to_f16_kernel(const float* __restrict__ in,
                                      _Float16* __restrict__ out, long long n) {
    long long i = (long long)blockIdx.x * blockDim.x + threadIdx.x;
    long long stride = (long long)gridDim.x * blockDim.x;
    for (; i < n; i += stride) out[i] = (_Float16)in[i];
}

// ---------------------------------------------------------- weight packing
// Pack W[K][N] (f32 row-major) into fragment-major f16:
//   P[(((tn*KT + kbi)*32 + lane)*16) + hh] = W[k][tn*16 + lane%16]
// with k = kbi*32 + (hh&7) + 8*(lane/16) + 16*(hh/8)
// so a wave's B fragment for tile (tn, kbi) is 32 contiguous bytes per lane.
__global__ void pack_weight_kernel(const float* __restrict__ W,
                                   _Float16* __restrict__ P, int K, int N) {
    int KT = K >> 5;
    long long total = (long long)K * N;
    long long i = (long long)blockIdx.x * blockDim.x + threadIdx.x;
    long long stride = (long long)gridDim.x * blockDim.x;
    for (; i < total; i += stride) {
        int hh   = (int)(i & 15);
        int lane = (int)((i >> 4) & 31);
        long long t = i >> 9;
        int kbi = (int)(t % KT);
        int tn  = (int)(t / KT);
        int g   = lane >> 4;
        int n   = tn * 16 + (lane & 15);
        int k   = kbi * 32 + (hh & 7) + 8 * g + ((hh >> 3) << 4);
        P[i] = (_Float16)W[(long long)k * N + n];
    }
}

// ----------------------------------------------------- scatter-sum (layer 1)
__global__ void aggregate_f32_kernel(const float* __restrict__ x,
                                     const int* __restrict__ ei,
                                     float* __restrict__ agg,
                                     float* __restrict__ deg, int D) {
    int d = threadIdx.x;
    for (int e = blockIdx.x; e < N_EDGES; e += gridDim.x) {
        int en = e + gridDim.x;
        if (en < N_EDGES)   // prefetch next gather row (global_prefetch_b8)
            __builtin_prefetch(&x[(long long)ei[en] * D + d], 0, 1);
        int src = ei[e];
        int dst = ei[N_EDGES + e];
        atomicAdd(&agg[(long long)dst * D + d], x[(long long)src * D + d]);
        if (d == 0) atomicAdd(&deg[dst], 1.0f);
    }
}

// ----------------------------------------------------- scatter-sum (layer 2)
__global__ void aggregate_f16_kernel(const _Float16* __restrict__ h,
                                     const int* __restrict__ ei,
                                     float* __restrict__ agg, int D) {
    int d = threadIdx.x;
    for (int e = blockIdx.x; e < N_EDGES; e += gridDim.x) {
        int en = e + gridDim.x;
        if (en < N_EDGES)
            __builtin_prefetch(&h[(long long)ei[en] * D + d], 0, 1);
        int src = ei[e];
        int dst = ei[N_EDGES + e];
        atomicAdd(&agg[(long long)dst * D + d], (float)h[(long long)src * D + d]);
    }
}

// ------------------------------------------------ mean = agg/deg, store f16
__global__ void mean_fin_kernel(const float* __restrict__ agg,
                                const float* __restrict__ deg,
                                _Float16* __restrict__ out, int D, long long n) {
    long long i = (long long)blockIdx.x * blockDim.x + threadIdx.x;
    long long stride = (long long)gridDim.x * blockDim.x;
    for (; i < n; i += stride) {
        long long node = i / D;
        float dv = deg[node];
        out[i] = (_Float16)(agg[i] / fmaxf(dv, 1.0f));
    }
}

// --------------------------------------------------------------- WMMA GEMM
// C[M,N] = relu?(A0[M,K]@W0 + A1[M,K]@W1 + bias), W* pre-packed fragment-major.
// One wave32 owns a full 16-row x N strip: NT f32 accumulator tiles in VGPRs,
// A fragments loaded once per K-step and reused across all N tiles.
__device__ __forceinline__ void load_fragA(FragH& f, const _Float16* __restrict__ A,
                                           int m, int K, int kb, int g) {
    const _Float16* base = A + (long long)m * K + kb + 8 * g;
    f.q[0] = *(const v4u*)(base);        // halves 0..7  : K = kb+8g   .. +7
    f.q[1] = *(const v4u*)(base + 16);   // halves 8..15 : K = kb+16+8g.. +7
}

__device__ __forceinline__ void load_fragP(FragH& f, const _Float16* __restrict__ P,
                                           long long elem_base) {
    f.q[0] = *(const v4u*)(P + elem_base);
    f.q[1] = *(const v4u*)(P + elem_base + 8);
}

template <int K, int NT>
__global__ void gemm_two_wmma_kernel(const _Float16* __restrict__ A0,
                                     const _Float16* __restrict__ P0,
                                     const _Float16* __restrict__ A1,
                                     const _Float16* __restrict__ P1,
                                     const float* __restrict__ bias,
                                     float* __restrict__ outF,
                                     _Float16* __restrict__ outH,
                                     int M, int relu) {
    constexpr int N  = NT * 16;
    constexpr int KT = K / 32;
    int wave = threadIdx.x >> 5;
    int lane = threadIdx.x & 31;
    int tm = blockIdx.x * (blockDim.x >> 5) + wave;
    if (tm >= (M >> 4)) return;           // wave-uniform: EXEC stays all-1s

    int g   = lane >> 4;
    int l16 = lane & 15;
    int row0 = tm << 4;
    int m = row0 + l16;

    v8f acc[NT];
#pragma unroll
    for (int tn = 0; tn < NT; ++tn) {
        float bv = bias[tn * 16 + l16];
#pragma unroll
        for (int r = 0; r < 8; ++r) acc[tn][r] = bv;
    }

#pragma unroll
    for (int kbi = 0; kbi < KT; ++kbi) {
        FragH a0, a1;
        load_fragA(a0, A0, m, K, kbi * 32, g);
        load_fragA(a1, A1, m, K, kbi * 32, g);
#pragma unroll
        for (int tn = 0; tn < NT; ++tn) {
            long long base = ((long long)(tn * KT + kbi) * 32 + lane) * 16;
            FragH b0, b1;
            load_fragP(b0, P0, base);
            load_fragP(b1, P1, base);
            acc[tn] = __builtin_amdgcn_wmma_f32_16x16x32_f16(
                false, a0.v, false, b0.v, (short)0, acc[tn], false, false);
            acc[tn] = __builtin_amdgcn_wmma_f32_16x16x32_f16(
                false, a1.v, false, b1.v, (short)0, acc[tn], false, false);
        }
    }

#pragma unroll
    for (int tn = 0; tn < NT; ++tn) {
#pragma unroll
        for (int r = 0; r < 8; ++r) {
            float v = acc[tn][r];
            if (relu) v = fmaxf(v, 0.0f);
            int row = row0 + r + (g << 3);
            long long idx = (long long)row * N + tn * 16 + l16;
            if (outF) outF[idx] = v;
            if (outH) outH[idx] = (_Float16)v;
        }
    }
}

// ------------------------------------------------------------------ decoder
__global__ void decoder_kernel(const float* __restrict__ z,
                               const int* __restrict__ eli,
                               const float* __restrict__ Wd1, const float* __restrict__ bd1,
                               const float* __restrict__ Wd2, const float* __restrict__ bd2,
                               const float* __restrict__ Wd3, const float* __restrict__ bd3,
                               float* __restrict__ out) {
    __shared__ float sW1[128 * 32];
    __shared__ float sW2[32 * 16];
    __shared__ float sW3[16];
    __shared__ float sb1[32];
    __shared__ float sb2[16];
    __shared__ float sb3;

    for (int t = threadIdx.x; t < 128 * 32; t += blockDim.x) sW1[t] = Wd1[t];
    for (int t = threadIdx.x; t < 32 * 16;  t += blockDim.x) sW2[t] = Wd2[t];
    if (threadIdx.x < 16) { sW3[threadIdx.x] = Wd3[threadIdx.x]; sb2[threadIdx.x] = bd2[threadIdx.x]; }
    if (threadIdx.x < 32) sb1[threadIdx.x] = bd1[threadIdx.x];
    if (threadIdx.x == 0) sb3 = bd3[0];
    __syncthreads();

    long long stride = (long long)gridDim.x * blockDim.x;
    for (long long p = (long long)blockIdx.x * blockDim.x + threadIdx.x;
         p < N_PRED; p += stride) {
        int i = eli[p];
        int j = eli[N_PRED + p];
        const float* zi = z + (long long)i * D_OUT;
        const float* zj = z + (long long)j * D_OUT;

        float f1[32];
#pragma unroll
        for (int k = 0; k < 32; ++k) f1[k] = sb1[k];
        for (int d = 0; d < D_OUT; ++d) {
            float v = zi[d];
            const float* w = sW1 + d * 32;
#pragma unroll
            for (int k = 0; k < 32; ++k) f1[k] += v * w[k];
        }
        for (int d = 0; d < D_OUT; ++d) {
            float v = zj[d];
            const float* w = sW1 + (D_OUT + d) * 32;
#pragma unroll
            for (int k = 0; k < 32; ++k) f1[k] += v * w[k];
        }
        float f2[16];
#pragma unroll
        for (int q = 0; q < 16; ++q) f2[q] = sb2[q];
#pragma unroll
        for (int k = 0; k < 32; ++k) {
            float v = fmaxf(f1[k], 0.0f);
            const float* w = sW2 + k * 16;
#pragma unroll
            for (int q = 0; q < 16; ++q) f2[q] += v * w[q];
        }
        float o = sb3;
#pragma unroll
        for (int q = 0; q < 16; ++q) o += fmaxf(f2[q], 0.0f) * sW3[q];
        out[p] = o;
    }
}

// ------------------------------------------------------------------- launch
extern "C" void kernel_launch(void* const* d_in, const int* in_sizes, int n_in,
                              void* d_out, int out_size, void* d_ws, size_t ws_size,
                              hipStream_t stream) {
    const float* x   = (const float*)d_in[0];
    const int*   ei  = (const int*)d_in[1];
    const int*   eli = (const int*)d_in[2];
    const float* W1l = (const float*)d_in[3];
    const float* b1l = (const float*)d_in[4];
    const float* W1r = (const float*)d_in[5];
    const float* W2l = (const float*)d_in[6];
    const float* b2l = (const float*)d_in[7];
    const float* W2r = (const float*)d_in[8];
    const float* Wd1 = (const float*)d_in[9];
    const float* bd1 = (const float*)d_in[10];
    const float* Wd2 = (const float*)d_in[11];
    const float* bd2 = (const float*)d_in[12];
    const float* Wd3 = (const float*)d_in[13];
    const float* bd3 = (const float*)d_in[14];
    float* out = (float*)d_out;

    // ---- workspace carve-up (all offsets multiples of 16 bytes)
    char* ws = (char*)d_ws;
    size_t off = 0;
    float* deg  = (float*)(ws + off);                 off += (size_t)N_NODES * 4;
    float* agg1 = (float*)(ws + off);  size_t agg1_off = off; off += (size_t)N_NODES * D_IN  * 4;
    float* agg2 = (float*)(ws + off);                 off += (size_t)N_NODES * D_HID * 4;
    size_t zero_floats = off / 4;                     // deg|agg1|agg2 contiguous
    _Float16* mean1h = (_Float16*)(ws + off);         off += (size_t)N_NODES * D_IN  * 2;
    _Float16* mean2h = (_Float16*)(ws + off);         off += (size_t)N_NODES * D_HID * 2;
    _Float16* xh     = (_Float16*)(ws + off);         off += (size_t)N_NODES * D_IN  * 2;
    _Float16* hh     = (_Float16*)(ws + off);         off += (size_t)N_NODES * D_HID * 2;
    _Float16* P1l    = (_Float16*)(ws + off);         off += (size_t)D_IN  * D_HID * 2;
    _Float16* P1r    = (_Float16*)(ws + off);         off += (size_t)D_IN  * D_HID * 2;
    _Float16* P2l    = (_Float16*)(ws + off);         off += (size_t)D_HID * D_OUT * 2;
    _Float16* P2r    = (_Float16*)(ws + off);         off += (size_t)D_HID * D_OUT * 2;
    float* z = (float*)(ws + agg1_off);               // alias: agg1 dead before z written
    (void)ws_size; (void)in_sizes; (void)n_in; (void)out_size;

    // 1) zero accumulators
    zero_f32_kernel<<<2048, 256, 0, stream>>>((float*)ws, (long long)zero_floats);

    // 2) stage x in f16; pack weights fragment-major f16
    cvt_f32_to_f16_kernel<<<2048, 256, 0, stream>>>(x, xh, (long long)N_NODES * D_IN);
    pack_weight_kernel<<<32, 256, 0, stream>>>(W1l, P1l, D_IN,  D_HID);
    pack_weight_kernel<<<32, 256, 0, stream>>>(W1r, P1r, D_IN,  D_HID);
    pack_weight_kernel<<<32, 256, 0, stream>>>(W2l, P2l, D_HID, D_OUT);
    pack_weight_kernel<<<32, 256, 0, stream>>>(W2r, P2r, D_HID, D_OUT);

    // 3) layer-1 scatter mean
    aggregate_f32_kernel<<<65536, D_IN, 0, stream>>>(x, ei, agg1, deg, D_IN);
    mean_fin_kernel<<<2048, 256, 0, stream>>>(agg1, deg, mean1h, D_IN,
                                              (long long)N_NODES * D_IN);

    // 4) layer-1 GEMM: h = relu(mean1@W1l + x@W1r + b1l) -> f16
    {
        int strips = N_NODES / 16;                    // 3125
        int blocks = (strips + 7) / 8;
        gemm_two_wmma_kernel<D_IN, D_HID / 16><<<blocks, 256, 0, stream>>>(
            mean1h, P1l, xh, P1r, b1l, nullptr, hh, N_NODES, 1);
    }

    // 5) layer-2 scatter mean (same deg)
    aggregate_f16_kernel<<<65536, D_HID, 0, stream>>>(hh, ei, agg2, D_HID);
    mean_fin_kernel<<<2048, 256, 0, stream>>>(agg2, deg, mean2h, D_HID,
                                              (long long)N_NODES * D_HID);

    // 6) layer-2 GEMM: z = mean2@W2l + h@W2r + b2l -> f32
    {
        int strips = N_NODES / 16;
        int blocks = (strips + 7) / 8;
        gemm_two_wmma_kernel<D_HID, D_OUT / 16><<<blocks, 256, 0, stream>>>(
            mean2h, P2l, hh, P2r, b2l, z, nullptr, N_NODES, 0);
    }

    // 7) decoder MLP over 400k edge pairs
    decoder_kernel<<<(N_PRED + 255) / 256, 256, 0, stream>>>(
        z, eli, Wd1, bd1, Wd2, bd2, Wd3, bd3, out);
}
